// PointNetSetAbstraction_33655363732049
// MI455X (gfx1250) — compile-verified, matrix operand order checked by reading, not run on hardware
//
#include <hip/hip_runtime.h>

#define BATCH    8
#define NPOINTS  16384
#define NPT      1024
#define KSAMP    32
#define RAD2     0.04f
#define BN_EPS   1e-5f

typedef _Float16 half8  __attribute__((ext_vector_type(8)));
typedef _Float16 half16 __attribute__((ext_vector_type(16)));
typedef float    float8 __attribute__((ext_vector_type(8)));

static constexpr size_t ROWS   = (size_t)BATCH * NPT * KSAMP;  // 262144
static constexpr int    MTILES = (int)(ROWS / 16);             // 16384

__device__ __forceinline__ void atomic_addf(float* p, float v) {
    __hip_atomic_fetch_add(p, v, __ATOMIC_RELAXED, __HIP_MEMORY_SCOPE_AGENT);
}

// ---------------- FPS: one block per batch, distances live in VGPRs ----------------
__global__ void fps_kernel(const float* __restrict__ xyz, int* __restrict__ fidx) {
    const int b   = blockIdx.x;
    const int tid = threadIdx.x;                  // 256 threads, 64 points each
    const float* X = xyz + (size_t)b * NPOINTS * 3;
    __shared__ float s_best[256];
    __shared__ int   s_bidx[256];
    __shared__ float s_c[3];
    __shared__ int   s_cur;

    float dist[64];
#pragma unroll
    for (int i = 0; i < 64; ++i) dist[i] = 1e10f;
    if (tid == 0) s_cur = 0;
    __syncthreads();

    for (int it = 0; it < NPT; ++it) {
        if (tid == 0) {
            int cur = s_cur;
            fidx[b * NPT + it] = cur;
            s_c[0] = X[cur * 3 + 0];
            s_c[1] = X[cur * 3 + 1];
            s_c[2] = X[cur * 3 + 2];
        }
        __syncthreads();
        const float cx = s_c[0], cy = s_c[1], cz = s_c[2];
        float best = -1.0f; int bi = 0;
#pragma unroll 4
        for (int i = 0; i < 64; ++i) {
            const int j = tid * 64 + i;
            const float dx = X[j * 3 + 0] - cx;
            const float dy = X[j * 3 + 1] - cy;
            const float dz = X[j * 3 + 2] - cz;
            const float d  = dx * dx + dy * dy + dz * dz;
            const float nd = fminf(dist[i], d);
            dist[i] = nd;
            if (nd > best) { best = nd; bi = j; }
        }
        s_best[tid] = best; s_bidx[tid] = bi;
        __syncthreads();
        for (int off = 128; off > 0; off >>= 1) {
            if (tid < off && s_best[tid + off] > s_best[tid]) {
                s_best[tid] = s_best[tid + off];
                s_bidx[tid] = s_bidx[tid + off];
            }
            __syncthreads();
        }
        if (tid == 0) s_cur = s_bidx[0];
        __syncthreads();
    }
}

// ------------- ball query + gather + feature build (wave32 per centroid) -------------
__global__ void ballgroup_kernel(const float* __restrict__ xyz,
                                 const float* __restrict__ pts,
                                 const int* __restrict__ fidx,
                                 float* __restrict__ out_xyz,
                                 _Float16* __restrict__ A0) {
    __shared__ int idxbuf[8][KSAMP];
    const int wv   = threadIdx.x >> 5;
    const int lane = threadIdx.x & 31;
    const int w    = (int)((blockIdx.x * blockDim.x + threadIdx.x) >> 5);
    if (w >= BATCH * NPT) return;
    const int b = w >> 10;                                   // NPT == 1024
    const float* X = xyz + (size_t)b * NPOINTS * 3;
    const float* P = pts + (size_t)b * NPOINTS * 3;
    const int ci = fidx[w];
    const float cx = X[ci * 3 + 0], cy = X[ci * 3 + 1], cz = X[ci * 3 + 2];
    if (lane == 0) {
        out_xyz[w * 3 + 0] = cx;
        out_xyz[w * 3 + 1] = cy;
        out_xyz[w * 3 + 2] = cz;
    }
    int cnt = 0;
    for (int base = 0; base < NPOINTS && cnt < KSAMP; base += 32) {
        const int j = base + lane;
        const float dx = X[j * 3 + 0] - cx;
        const float dy = X[j * 3 + 1] - cy;
        const float dz = X[j * 3 + 2] - cz;
        const bool in = (dx * dx + dy * dy + dz * dz) <= RAD2;
        const unsigned mask = (unsigned)__ballot(in);
        if (in) {
            const int pos = cnt + __popc(mask & ((1u << lane) - 1u));
            if (pos < KSAMP) idxbuf[wv][pos] = j;
        }
        cnt += __popc(mask);
    }
    if (cnt > KSAMP) cnt = KSAMP;
    const int gi = idxbuf[wv][lane < cnt ? lane : 0];        // pad with first in-ball idx
    const size_t r = (size_t)w * KSAMP + lane;
    _Float16 row[8] = {
        (_Float16)(X[gi * 3 + 0] - cx), (_Float16)(X[gi * 3 + 1] - cy),
        (_Float16)(X[gi * 3 + 2] - cz),
        (_Float16)P[gi * 3 + 0], (_Float16)P[gi * 3 + 1], (_Float16)P[gi * 3 + 2],
        (_Float16)0.f, (_Float16)0.f };
    half8 z = {};
    half8* dst = (half8*)(A0 + r * 32);                      // K padded 6 -> 32
    dst[0] = *(half8*)row;
    dst[1] = z; dst[2] = z; dst[3] = z;
}

// ---------------- weight conversion fp32 -> f16, K padded for layer 0 ----------------
__global__ void prep_weights(const float* __restrict__ w0, const float* __restrict__ w1,
                             const float* __restrict__ w2,
                             _Float16* __restrict__ w0p, _Float16* __restrict__ w1p,
                             _Float16* __restrict__ w2p) {
    const int t = blockIdx.x * blockDim.x + threadIdx.x;
    if (t < 64 * 32) {
        const int n = t >> 5, k = t & 31;
        w0p[t] = (_Float16)(k < 6 ? w0[n * 6 + k] : 0.f);
    }
    if (t < 64 * 64)  w1p[t] = (_Float16)w1[t];
    if (t < 128 * 64) w2p[t] = (_Float16)w2[t];
}

__global__ void zero_sums(float* __restrict__ s) {
    const int t = threadIdx.x;
    s[t] = 0.f; s[t + 256] = 0.f; s[t + 512] = 0.f;          // 3 layers x (sum,sumsq) x 128
}

// ---------------- WMMA GEMM: Y[ROWS,NT*16] = A[ROWS,K] * W[NT*16,K]^T ----------------
// 8 waves/block, one M-tile per wave, all NT N-tiles per wave (A fragment reuse).
// Weights staged once per block into LDS; B fragments come from ds_load_b128.
template <int K, int NT>
__global__ void gemm_wmma(const _Float16* __restrict__ A, const _Float16* __restrict__ W,
                          float* __restrict__ Y, float* __restrict__ sums) {
    constexpr int N = NT * 16;
    __shared__ _Float16 Wlds[N * K];                         // <= 16 KB

    // cooperative weight stage (vectorized b128)
    {
        constexpr int CH = (N * K) / 8;                      // half8 chunks
        half8* d = (half8*)Wlds;
        const half8* s = (const half8*)W;
        for (int i = threadIdx.x; i < CH; i += 256) d[i] = s[i];
    }
    __syncthreads();

    const int lane = threadIdx.x & 31;
    const int hl   = lane >> 4;                              // half-wave select
    const int idx  = lane & 15;
    const int mt   = (int)blockIdx.x * 8 + (threadIdx.x >> 5);

    float8 acc[NT];
#pragma unroll
    for (int n = 0; n < NT; ++n) acc[n] = float8{};

    // ISA 16-bit A layout: two contiguous 8-half chunks at 8*hl and 16+8*hl
    const _Float16* arow = A + (size_t)(mt * 16 + idx) * K + 8 * hl;
#pragma unroll
    for (int kb = 0; kb < K; kb += 32) {
        union { half16 v; half8 h[2]; } a;
        a.h[0] = *(const half8*)(arow + kb);
        a.h[1] = *(const half8*)(arow + kb + 16);
#pragma unroll
        for (int n = 0; n < NT; ++n) {
            union { half16 v; half8 h[2]; } bfr;
            const _Float16* brow = Wlds + (size_t)(n * 16 + idx) * K + 8 * hl + kb;
            bfr.h[0] = *(const half8*)(brow);
            bfr.h[1] = *(const half8*)(brow + 16);
            acc[n] = __builtin_amdgcn_wmma_f32_16x16x32_f16(false, a.v, false, bfr.v,
                                                            (short)0, acc[n], false, false);
        }
    }
#pragma unroll
    for (int n = 0; n < NT; ++n) {
        const int col = n * 16 + idx;
        float s = 0.f, s2 = 0.f;
#pragma unroll
        for (int r = 0; r < 8; ++r) {
            const float v = acc[n][r];
            Y[(size_t)(mt * 16 + 8 * hl + r) * N + col] = v;
            s += v; s2 += v * v;
        }
        atomic_addf(&sums[col], s);
        atomic_addf(&sums[128 + col], s2);
    }
}

// ---------------- BN stat finalize: scale/shift per channel ----------------
__global__ void bn_finalize(const float* __restrict__ sums, const float* __restrict__ g,
                            const float* __restrict__ be, float* __restrict__ ss, int C) {
    const int c = threadIdx.x;
    if (c < C) {
        const float inv = 1.0f / (float)ROWS;
        const float mu  = sums[c] * inv;
        const float var = sums[128 + c] * inv - mu * mu;
        const float sc  = g[c] * rsqrtf(var + BN_EPS);
        ss[c]       = sc;
        ss[128 + c] = be[c] - mu * sc;
    }
}

// ---------------- BN + ReLU, emit f16 activations for next WMMA layer (C==64) ----------------
__global__ void bn_apply_relu(const float* __restrict__ Y, const float* __restrict__ ss,
                              _Float16* __restrict__ Aout) {
    const size_t i = (size_t)blockIdx.x * blockDim.x + threadIdx.x;
    const int c = (int)(i & 63);
    float v = Y[i] * ss[c] + ss[128 + c];
    v = v > 0.f ? v : 0.f;
    Aout[i] = (_Float16)v;
}

// ---------------- fused BN + ReLU + max over KSAMP (layer 3, C==128) ----------------
__global__ void maxpool_kernel(const float* __restrict__ Y, const float* __restrict__ ss,
                               float* __restrict__ out) {
    const int t = blockIdx.x * blockDim.x + threadIdx.x;     // BATCH*NPT*128
    const int c = t & 127;
    const int w = t >> 7;
    const float sc = ss[c], sh = ss[128 + c];
    float m = 0.f;                                           // relu() >= 0
    const size_t base = (size_t)w * KSAMP * 128 + c;
#pragma unroll 8
    for (int k = 0; k < KSAMP; ++k) {
        float v = Y[base + (size_t)k * 128] * sc + sh;
        v = v > 0.f ? v : 0.f;
        m = v > m ? v : m;
    }
    out[t] = m;
}

extern "C" void kernel_launch(void* const* d_in, const int* in_sizes, int n_in,
                              void* d_out, int out_size, void* d_ws, size_t ws_size,
                              hipStream_t stream) {
    (void)in_sizes; (void)n_in; (void)out_size; (void)ws_size;
    const float* xyz = (const float*)d_in[0];
    const float* pts = (const float*)d_in[1];
    const float* w0  = (const float*)d_in[2];
    const float* g0  = (const float*)d_in[4];
    const float* be0 = (const float*)d_in[5];
    const float* w1  = (const float*)d_in[6];
    const float* g1  = (const float*)d_in[8];
    const float* be1 = (const float*)d_in[9];
    const float* w2  = (const float*)d_in[10];
    const float* g2  = (const float*)d_in[12];
    const float* be2 = (const float*)d_in[13];
    float* out = (float*)d_out;

    char* ws = (char*)d_ws;
    int*        fidx = (int*)      (ws + 0);                 // 8192 ints
    float*      sums = (float*)    (ws + 32768);             // 3*256 floats
    float*      ss   = (float*)    (ws + 36864);             // 3*256 floats
    _Float16*   w0p  = (_Float16*) (ws + 40960);             // 64x32
    _Float16*   w1p  = (_Float16*) (ws + 45056);             // 64x64
    _Float16*   w2p  = (_Float16*) (ws + 53248);             // 128x64
    _Float16*   A0   = (_Float16*) (ws + (1ull  << 20));     // ROWS x 32 f16 (16 MiB)
    _Float16*   A1   = (_Float16*) (ws + (17ull << 20));     // ROWS x 64 f16 (32 MiB)
    _Float16*   A2   = (_Float16*) (ws + (49ull << 20));     // ROWS x 64 f16 (32 MiB)
    float*      Y    = (float*)    (ws + (81ull << 20));     // ROWS x 128 f32 (128 MiB)

    zero_sums<<<1, 256, 0, stream>>>(sums);
    prep_weights<<<32, 256, 0, stream>>>(w0, w1, w2, w0p, w1p, w2p);
    fps_kernel<<<BATCH, 256, 0, stream>>>(xyz, fidx);
    ballgroup_kernel<<<(BATCH * NPT * 32) / 256, 256, 0, stream>>>(xyz, pts, fidx, out, A0);

    // layer 0: K=32 (padded from 6), N=64 : 16384 M-tiles / 8 waves per block
    gemm_wmma<32, 4><<<MTILES / 8, 256, 0, stream>>>(A0, w0p, Y, sums + 0);
    bn_finalize<<<1, 64, 0, stream>>>(sums + 0, g0, be0, ss + 0, 64);
    bn_apply_relu<<<(int)(ROWS * 64 / 256), 256, 0, stream>>>(Y, ss + 0, A1);

    // layer 1: K=64, N=64
    gemm_wmma<64, 4><<<MTILES / 8, 256, 0, stream>>>(A1, w1p, Y, sums + 256);
    bn_finalize<<<1, 64, 0, stream>>>(sums + 256, g1, be1, ss + 256, 64);
    bn_apply_relu<<<(int)(ROWS * 64 / 256), 256, 0, stream>>>(Y, ss + 256, A2);

    // layer 2: K=64, N=128
    gemm_wmma<64, 8><<<MTILES / 8, 256, 0, stream>>>(A2, w2p, Y, sums + 512);
    bn_finalize<<<1, 128, 0, stream>>>(sums + 512, g2, be2, ss + 512, 128);

    maxpool_kernel<<<(BATCH * NPT * 128) / 256, 256, 0, stream>>>(Y, ss + 512,
                                                                  out + BATCH * NPT * 3);
}
